// GridDVAE_16578573762806
// MI455X (gfx1250) — matrix operand order repaired and checked by reading, not compiled, not run on hardware
//
#include <hip/hip_runtime.h>
#include <hip/hip_bf16.h>
#include <math.h>

typedef __bf16 bf16;
typedef bf16  bf16x16 __attribute__((ext_vector_type(16)));
typedef bf16  bf16x8  __attribute__((ext_vector_type(8)));
typedef float floatx8 __attribute__((ext_vector_type(8)));

#define D_DIM  1024
#define BATCH_N 32
#define CODES  512

enum { OUT_BF16 = 0, OUT_F32 = 1, OUT_BF16_T = 2 };

// ---------------- elementwise converts ----------------
__global__ __launch_bounds__(256)
void convert_f32_bf16_kernel(const float* __restrict__ in, bf16* __restrict__ out, long n) {
  long i = (long)blockIdx.x * 256 + threadIdx.x;
  if (i < n) out[i] = (bf16)in[i];
}

// out[c*R + r] = in[r*C + c]  (W -> W^T, fp32 -> bf16)
__global__ __launch_bounds__(256)
void transpose_f32_bf16_kernel(const float* __restrict__ in, bf16* __restrict__ out, int R, int C) {
  long i = (long)blockIdx.x * 256 + threadIdx.x;
  if (i < (long)R * C) {
    int r = (int)(i / C), c = (int)(i % C);
    out[(long)c * R + r] = (bf16)in[i];
  }
}

// ---------------- bf16 WMMA GEMM, 64x64 tile per wave (4x4 register blocking) ----
// D[M,N] = scale * (A[M,Kred] * Bt[N,Kred]^T), fp32 accumulate.
// grid = (M/64, N/64, nbatch), block = 1 wave32.
// Per k-step of 32: 4 A-fragments + 4 B-fragments feed 16 V_WMMA_F32_16X16X32_BF16.
// Fragment packing follows CDNA5 ISA 7.12.2 (16-bit A 16x32, B 32x16, f32 C 16x16).
__global__ __launch_bounds__(32)
void gemm_bf16_wmma_kernel(const bf16* __restrict__ A, long lda, long strideA,
                           const bf16* __restrict__ Bt, long ldb, long strideB,
                           void* __restrict__ Cout, long ldc, long strideC,
                           int Kred, float scale, int outMode) {
  const int tm = blockIdx.x, tn = blockIdx.y, b = blockIdx.z;
  const int lane = threadIdx.x;           // 0..31, wave32
  const int mr   = lane & 15;             // A row / B col / C col within a 16-tile
  const int hi   = lane >> 4;             // lane half select

  const bf16* __restrict__ Ab = A  + (long)b * strideA + (long)tm * 64 * lda;
  const bf16* __restrict__ Bb = Bt + (long)b * strideB + (long)tn * 64 * ldb;

  const long aoff = (long)mr * lda + (hi << 3);   // A: k-base 0 or 8 (upper half at +16)
  const long boff = (long)mr * ldb + (hi << 3);   // B: k-base 0 or 8 (upper half at +16)

  floatx8 acc[4][4];
#pragma unroll
  for (int mi = 0; mi < 4; ++mi)
#pragma unroll
    for (int ni = 0; ni < 4; ++ni) acc[mi][ni] = floatx8{};

#pragma unroll 1
  for (int k0 = 0; k0 < Kred; k0 += 32) {
    bf16x16 af[4], bg[4];
#pragma unroll
    for (int i = 0; i < 4; ++i) {
      const bf16* ap = Ab + (long)i * 16 * lda + aoff + k0;
      const bf16* bp = Bb + (long)i * 16 * ldb + boff + k0;
      bf16x8 a0 = *(const bf16x8*)(ap);        // A K = kb+0..7
      bf16x8 a1 = *(const bf16x8*)(ap + 16);   // A K = kb+16..23
      bf16x8 b0 = *(const bf16x8*)(bp);        // B K = kb+0..7
      bf16x8 b1 = *(const bf16x8*)(bp + 16);   // B K = kb+16..23
#pragma unroll
      for (int j = 0; j < 8; ++j) {
        af[i][j] = a0[j]; af[i][j + 8] = a1[j];
        bg[i][j] = b0[j]; bg[i][j + 8] = b1[j];
      }
    }
#pragma unroll
    for (int mi = 0; mi < 4; ++mi)
#pragma unroll
      for (int ni = 0; ni < 4; ++ni)
        acc[mi][ni] = __builtin_amdgcn_wmma_f32_16x16x32_bf16(
            false, af[mi], false, bg[ni], (short)0, acc[mi][ni], false, false);
  }

  const int mb = hi << 3;   // C: VGPR r -> row r (lanes 0-15) / row 8+r (lanes 16-31)
  if (outMode == OUT_F32) {
    float* Cb = (float*)Cout + (long)b * strideC;
#pragma unroll
    for (int mi = 0; mi < 4; ++mi)
#pragma unroll
      for (int ni = 0; ni < 4; ++ni) {
        float* Cp = Cb + (long)(tm * 64 + mi * 16 + mb) * ldc + tn * 64 + ni * 16 + mr;
#pragma unroll
        for (int r = 0; r < 8; ++r) Cp[(long)r * ldc] = acc[mi][ni][r] * scale;
      }
  } else if (outMode == OUT_BF16) {
    bf16* Cb = (bf16*)Cout + (long)b * strideC;
#pragma unroll
    for (int mi = 0; mi < 4; ++mi)
#pragma unroll
      for (int ni = 0; ni < 4; ++ni) {
        bf16* Cp = Cb + (long)(tm * 64 + mi * 16 + mb) * ldc + tn * 64 + ni * 16 + mr;
#pragma unroll
        for (int r = 0; r < 8; ++r) Cp[(long)r * ldc] = (bf16)(acc[mi][ni][r] * scale);
      }
  } else { // OUT_BF16_T: store D^T, i.e. out[n*ldc + m] (8 contiguous bf16 per lane)
    bf16* Cb = (bf16*)Cout + (long)b * strideC;
#pragma unroll
    for (int mi = 0; mi < 4; ++mi)
#pragma unroll
      for (int ni = 0; ni < 4; ++ni) {
        bf16* Cp = Cb + (long)(tn * 64 + ni * 16 + mr) * ldc + tm * 64 + mi * 16 + mb;
#pragma unroll
        for (int r = 0; r < 8; ++r) Cp[r] = (bf16)(acc[mi][ni][r] * scale);
      }
  }
}

// ---------------- softmax over last dim (fp32 in, bf16 out) ----------------
__global__ __launch_bounds__(256)
void softmax_attn_kernel(const float* __restrict__ logits, bf16* __restrict__ attn, int S) {
  const long row = blockIdx.x;
  const float* __restrict__ in = logits + row * (long)S;
  bf16* __restrict__ out = attn + row * (long)S;
  __shared__ float buf[256];
  const int tid = threadIdx.x;
  float m = -1e30f;
  for (int i = tid; i < S; i += 256) m = fmaxf(m, in[i]);
  buf[tid] = m; __syncthreads();
  for (int s = 128; s > 0; s >>= 1) { if (tid < s) buf[tid] = fmaxf(buf[tid], buf[tid + s]); __syncthreads(); }
  m = buf[0]; __syncthreads();
  float sum = 0.f;
  for (int i = tid; i < S; i += 256) sum += __expf(in[i] - m);
  buf[tid] = sum; __syncthreads();
  for (int s = 128; s > 0; s >>= 1) { if (tid < s) buf[tid] += buf[tid + s]; __syncthreads(); }
  const float inv = 1.0f / buf[0];
  for (int i = tid; i < S; i += 256) out[i] = (bf16)(__expf(in[i] - m) * inv);
}

// ---------------- ||codebook_c||^2 ----------------
__global__ __launch_bounds__(256)
void cnorm_kernel(const float* __restrict__ cb, float* __restrict__ cnorm, int Dd) {
  const int c = blockIdx.x, tid = threadIdx.x;
  __shared__ float buf[256];
  const float* __restrict__ row = cb + (long)c * Dd;
  float s = 0.f;
  for (int i = tid; i < Dd; i += 256) { float v = row[i]; s += v * v; }
  buf[tid] = s; __syncthreads();
  for (int st = 128; st > 0; st >>= 1) { if (tid < st) buf[tid] += buf[tid + st]; __syncthreads(); }
  if (tid == 0) cnorm[c] = buf[0];
}

// ---------------- argmin over codes + fp32 gather (out = z_q) ----------------
__global__ __launch_bounds__(256)
void vq_argmin_gather_kernel(const float* __restrict__ dots, const float* __restrict__ cnorm,
                             const float* __restrict__ cb, float* __restrict__ out, int C, int Dd) {
  const long l = blockIdx.x;
  const int tid = threadIdx.x;
  __shared__ float bv[256];
  __shared__ int   bi[256];
  float best = 1e30f; int besti = 0;
  for (int c = tid; c < C; c += 256) {
    float d = cnorm[c] - 2.0f * dots[l * (long)C + c];   // |z|^2 constant per row
    if (d < best) { best = d; besti = c; }
  }
  bv[tid] = best; bi[tid] = besti; __syncthreads();
  for (int s = 128; s > 0; s >>= 1) {
    if (tid < s) {
      if (bv[tid + s] < bv[tid] || (bv[tid + s] == bv[tid] && bi[tid + s] < bi[tid])) {
        bv[tid] = bv[tid + s]; bi[tid] = bi[tid + s];
      }
    }
    __syncthreads();
  }
  const int c = bi[0];
  const float* __restrict__ src = cb + (long)c * Dd;
  float* __restrict__ dst = out + l * (long)Dd;
  for (int i = tid; i < Dd; i += 256) dst[i] = src[i];
}

// ---------------- host orchestration ----------------
extern "C" void kernel_launch(void* const* d_in, const int* in_sizes, int n_in,
                              void* d_out, int out_size, void* d_ws, size_t ws_size,
                              hipStream_t stream) {
  (void)in_sizes; (void)n_in; (void)out_size; (void)ws_size;
  const float* x        = (const float*)d_in[0];   // [32,1024,1024]
  const float* queries  = (const float*)d_in[1];   // [5,1024,1024]
  const float* wq       = (const float*)d_in[2];   // [5,1024,1024]
  const float* wk       = (const float*)d_in[3];
  const float* wv       = (const float*)d_in[4];
  const float* wo       = (const float*)d_in[5];
  const float* codebook = (const float*)d_in[6];   // [512,1024]
  float* out = (float*)d_out;                      // [32,64,1024]

  const int  Dd = D_DIM;
  const long DD = (long)Dd * Dd;
  static const int POOL[5] = {1024, 512, 256, 128, 64};

  char* ws = (char*)d_ws;
  size_t off = 0;
  auto take = [&](size_t bytes) -> void* {
    void* p = ws + off;
    off = (off + bytes + 255) & ~(size_t)255;
    return p;
  };
  const long maxSD = (long)BATCH_N * 1024 * Dd;        // 33,554,432 elems
  bf16*  hbA    = (bf16*)take(maxSD * sizeof(bf16));
  bf16*  hbB    = (bf16*)take(maxSD * sizeof(bf16));
  bf16*  kb     = (bf16*)take(maxSD * sizeof(bf16));   // also reused as `pooled`
  bf16*  vt     = (bf16*)take(maxSD * sizeof(bf16));   // v stored transposed [B,D,S]
  float* logits = (float*)take(maxSD * sizeof(float)); // [B,K,S] fp32
  bf16*  attn   = (bf16*)take(maxSD * sizeof(bf16));
  bf16*  qb     = (bf16*)take((size_t)1024 * Dd * sizeof(bf16));
  bf16*  qin    = (bf16*)take((size_t)1024 * Dd * sizeof(bf16));
  bf16*  wTq    = (bf16*)take((size_t)DD * sizeof(bf16));
  bf16*  wTk    = (bf16*)take((size_t)DD * sizeof(bf16));
  bf16*  wTv    = (bf16*)take((size_t)DD * sizeof(bf16));
  bf16*  wTo    = (bf16*)take((size_t)DD * sizeof(bf16));
  bf16*  cbb    = (bf16*)take((size_t)CODES * Dd * sizeof(bf16));
  float* cnorm  = (float*)take((size_t)CODES * sizeof(float));
  float* dots   = (float*)take((size_t)BATCH_N * 64 * CODES * sizeof(float));

  const float SCALE = 0.03125f;  // D^-0.5

  // h0 = bf16(x)
  convert_f32_bf16_kernel<<<dim3((unsigned)((maxSD + 255) / 256)), 256, 0, stream>>>(x, hbA, maxSD);

  bf16* hcur = hbA;
  bf16* hnext = hbB;
  for (int i = 0; i < 5; ++i) {
    const int S = (i == 0) ? 1024 : POOL[i - 1];
    const int K = POOL[i];
    const unsigned tgrid = (unsigned)((DD + 255) / 256);
    transpose_f32_bf16_kernel<<<tgrid, 256, 0, stream>>>(wq + (long)i * DD, wTq, Dd, Dd);
    transpose_f32_bf16_kernel<<<tgrid, 256, 0, stream>>>(wk + (long)i * DD, wTk, Dd, Dd);
    transpose_f32_bf16_kernel<<<tgrid, 256, 0, stream>>>(wv + (long)i * DD, wTv, Dd, Dd);
    transpose_f32_bf16_kernel<<<tgrid, 256, 0, stream>>>(wo + (long)i * DD, wTo, Dd, Dd);

    const long nq = (long)K * Dd;
    convert_f32_bf16_kernel<<<dim3((unsigned)((nq + 255) / 256)), 256, 0, stream>>>(
        queries + (long)i * 1024 * Dd, qin, nq);

    // q = queries_i[:K] @ wq_i            [K,D]
    gemm_bf16_wmma_kernel<<<dim3(K / 64, Dd / 64, 1), 32, 0, stream>>>(
        qin, Dd, 0, wTq, Dd, 0, qb, Dd, 0, Dd, 1.0f, OUT_BF16);
    // k = h @ wk_i                        [B,S,D]
    gemm_bf16_wmma_kernel<<<dim3(S / 64, Dd / 64, BATCH_N), 32, 0, stream>>>(
        hcur, Dd, (long)S * Dd, wTk, Dd, 0, kb, Dd, (long)S * Dd, Dd, 1.0f, OUT_BF16);
    // v^T = (h @ wv_i)^T                  [B,D,S]
    gemm_bf16_wmma_kernel<<<dim3(S / 64, Dd / 64, BATCH_N), 32, 0, stream>>>(
        hcur, Dd, (long)S * Dd, wTv, Dd, 0, vt, S, (long)Dd * S, Dd, 1.0f, OUT_BF16_T);
    // logits = scale * q @ k^T            [B,K,S] fp32
    gemm_bf16_wmma_kernel<<<dim3(K / 64, S / 64, BATCH_N), 32, 0, stream>>>(
        qb, Dd, 0, kb, Dd, (long)S * Dd, logits, S, (long)K * S, Dd, SCALE, OUT_F32);
    // attn = softmax(logits) -> bf16
    softmax_attn_kernel<<<dim3((unsigned)(BATCH_N * K)), 256, 0, stream>>>(logits, attn, S);
    // pooled = attn @ v                   [B,K,D]  (reuses kb buffer)
    bf16* pooled = kb;
    gemm_bf16_wmma_kernel<<<dim3(K / 64, Dd / 64, BATCH_N), 32, 0, stream>>>(
        attn, S, (long)K * S, vt, S, (long)Dd * S, pooled, Dd, (long)K * Dd, S, 1.0f, OUT_BF16);
    // h' = pooled @ wo_i  as one [B*K, D] GEMM
    gemm_bf16_wmma_kernel<<<dim3(BATCH_N * K / 64, Dd / 64, 1), 32, 0, stream>>>(
        pooled, Dd, 0, wTo, Dd, 0, hnext, Dd, 0, Dd, 1.0f, OUT_BF16);

    bf16* t = hcur; hcur = hnext; hnext = t;
  }

  // ---- VQ: out = codebook[argmin_c (|c|^2 - 2 z.c)] ----
  const long ncb = (long)CODES * Dd;
  convert_f32_bf16_kernel<<<dim3((unsigned)((ncb + 255) / 256)), 256, 0, stream>>>(codebook, cbb, ncb);
  cnorm_kernel<<<CODES, 256, 0, stream>>>(codebook, cnorm, Dd);
  // dots = z @ codebook^T : [2048, 512]; codebook's [C,D] layout is already the Bt layout
  gemm_bf16_wmma_kernel<<<dim3(BATCH_N * 64 / 64, CODES / 64, 1), 32, 0, stream>>>(
      hcur, Dd, 0, cbb, Dd, 0, dots, CODES, 0, Dd, 1.0f, OUT_F32);
  vq_argmin_gather_kernel<<<dim3(BATCH_N * 64), 256, 0, stream>>>(dots, cnorm, codebook, out, CODES, Dd);
}
